// AdaptiveScanMamba_50362786512972
// MI455X (gfx1250) — matrix operand order repaired and checked by reading, not compiled, not run on hardware
//
#include <hip/hip_runtime.h>
#include <hip/hip_bf16.h>

typedef __attribute__((ext_vector_type(16))) _Float16 v16h;
typedef __attribute__((ext_vector_type(8)))  float    v8f;

#define BATCH   8
#define NTOK    197
#define EDIM    256
#define DI      512
#define DSTATE  64
#define DTRANK  16
#define NPATCH  196
#define NPAD    200
#define MTOK    (BATCH*NTOK)    // 1576
#define MPATCH  (BATCH*NPATCH)  // 1568
#define MPAD    (BATCH*NPAD)    // 1600

__device__ __forceinline__ float gelu_f(float x)     { return 0.5f*x*(1.f+erff(x*0.70710678118654752f)); }
__device__ __forceinline__ float silu_f(float x)     { return x/(1.f+__expf(-x)); }
__device__ __forceinline__ float softplus_f(float x) { return (x>20.f)?x:log1pf(__expf(x)); }

// ---------------------------------------------------------------------------
// Generic WMMA GEMM:  C[M,N] (+)= act( A[M,K](lda) * Bw[N,K]^T + bias )
// 4 waves/block, 64x64 tile, BK=32 K-steps, f32->f16 staged in LDS,
// v_wmma_f32_16x16x32_f16 with f32 accumulation.
// Edge handling: row/col indices are CLAMPED on load (duplicated data lands in
// accumulator rows/cols that are never stored), so the load path has no
// lane-varying predication; the K-tail test is wave-uniform.
// ---------------------------------------------------------------------------
#define GBM 64
#define GBN 64
#define GBK 32
#define LDSK (GBK + 8)   // pad to spread LDS banks

__global__ __launch_bounds__(128) void gemm_wmma(
    const float* __restrict__ A, int lda,
    const float* __restrict__ Bw,          // [N,K] row-major (weight layout)
    const float* __restrict__ bias,        // [N] or nullptr
    float* __restrict__ C, int ldc,
    int M, int N, int K,
    int act,                               // 0 none, 1 gelu, 3 softplus
    int accumulate,                        // 0: C =, 1: C +=
    const float* __restrict__ rowScale)    // optional per-ROW scale [M]
{
    __shared__ _Float16 As[GBM][LDSK];
    __shared__ _Float16 Bs[GBN][LDSK];

    const int tid  = threadIdx.x;
    const int wave = tid >> 5;
    const int lane = tid & 31;
    const int m0 = blockIdx.y * GBM;
    const int n0 = blockIdx.x * GBN;

    v8f acc[4];
    #pragma unroll
    for (int nt = 0; nt < 4; ++nt) acc[nt] = 0.f;

    // Loader mapping: waves 0-1 -> k-half 0, waves 2-3 -> k-half 1 (wave-uniform c0)
    const int r  = tid & 63;            // 0..63 : tile row (A) / tile col (B)
    const int c0 = (tid >> 6) * 16;     // 0 or 16, uniform within a wave
    const int grA = min(m0 + r, M - 1); // clamped: no lane-varying predication
    const int gnB = min(n0 + r, N - 1);

    for (int k0 = 0; k0 < K; k0 += GBK) {
        const float* srcA = A  + (size_t)grA * lda + k0 + c0;
        const float* srcB = Bw + (size_t)gnB * K   + k0 + c0;
        if (k0 + c0 + 16 <= K) {        // wave-uniform branch
            #pragma unroll
            for (int j = 0; j < 16; ++j) As[r][c0 + j] = (_Float16)srcA[j];
            #pragma unroll
            for (int j = 0; j < 16; ++j) Bs[r][c0 + j] = (_Float16)srcB[j];
        } else {                        // K tail: per-j guards are scalar
            #pragma unroll
            for (int j = 0; j < 16; ++j) {
                float va = 0.f, vb = 0.f;
                if (k0 + c0 + j < K) { va = srcA[j]; vb = srcB[j]; }
                As[r][c0 + j] = (_Float16)va;
                Bs[r][c0 + j] = (_Float16)vb;
            }
        }
        if (k0 + GBK < K) {             // prefetch next K-chunk (global_prefetch_b8)
            __builtin_prefetch(srcA + GBK, 0, 3);
            __builtin_prefetch(srcB + GBK, 0, 3);
        }
        __syncthreads();

        // ---- A fragment (ISA 7.12.2: lanes0-15 K{0-7,16-23}, lanes16-31 K{8-15,24-31})
        const int row = wave * 16 + (lane & 15);
        const int kbA = (lane < 16) ? 0 : 8;
        v16h a;
        #pragma unroll
        for (int j = 0; j < 8; ++j) {
            a[j]     = As[row][kbA + j];
            a[8 + j] = As[row][kbA + 16 + j];
        }
        // ---- B fragments (lanes0-15 K{0-7,16-23}, lanes16-31 K{8-15,24-31}) + WMMA
        const int kbB = (lane < 16) ? 0 : 8;
        #pragma unroll
        for (int nt = 0; nt < 4; ++nt) {
            const int col = nt * 16 + (lane & 15);
            v16h b;
            #pragma unroll
            for (int j = 0; j < 8; ++j) { b[j] = Bs[col][kbB + j]; b[8 + j] = Bs[col][kbB + 16 + j]; }
            acc[nt] = __builtin_amdgcn_wmma_f32_16x16x32_f16(
                false, a, false, b, (short)0, acc[nt], false, false);
        }
        __syncthreads();
    }

    // ---- epilogue: C/D layout (VGPR v -> M = v + (lane<16?0:8), N = lane&15)
    const int rbase = m0 + wave * 16 + ((lane < 16) ? 0 : 8);
    const int cl = lane & 15;
    const bool interior = (m0 + GBM <= M) && (n0 + GBN <= N);  // block-uniform
    #pragma unroll
    for (int nt = 0; nt < 4; ++nt) {
        const int gc = n0 + nt * 16 + cl;
        #pragma unroll
        for (int v = 0; v < 8; ++v) {
            const int gr = rbase + v;
            if (interior || (gr < M && gc < N)) {
                float val = acc[nt][v];
                if (bias) val += bias[gc];
                if (act == 1)      val = gelu_f(val);
                else if (act == 3) val = softplus_f(val);
                if (rowScale) val *= rowScale[gr];
                float* p = &C[(size_t)gr * ldc + gc];
                if (accumulate) *p += val; else *p = val;
            }
        }
    }
}

// ---------------------------------------------------------------------------
// Small element-wise / scalar kernels
// ---------------------------------------------------------------------------
__global__ void k_weff(const float* __restrict__ pw, const float* __restrict__ cw,
                       float* __restrict__ weff) {
    int idx = blockIdx.x * 256 + threadIdx.x;
    if (idx >= EDIM * 768) return;
    int e = idx / 768, k = idx % 768;
    int ch = k >> 8, pq = k & 255;
    float s = 0.f;
    #pragma unroll
    for (int c = 0; c < 6; ++c) s += pw[((size_t)(e * 6 + c)) * 256 + pq] * cw[c * 3 + ch];
    weff[idx] = s;
}

__global__ void k_im2col(const float* __restrict__ x, float* __restrict__ col) {
    int idx = blockIdx.x * 256 + threadIdx.x;
    if (idx >= MPATCH * 768) return;
    int m = idx / 768, k = idx % 768;
    int ch = k >> 8, pq = k & 255, pp = pq >> 4, q = pq & 15;
    int b = m / NPATCH, pi = m % NPATCH, gi = pi / 14, gj = pi % 14;
    col[idx] = x[(((size_t)(b * 3 + ch)) * 224 + gi * 16 + pp) * 224 + gj * 16 + q];
}

__global__ void k_tok_assemble(const float* __restrict__ ptok, const float* __restrict__ cls,
                               const float* __restrict__ pos, float* __restrict__ tok) {
    int idx = blockIdx.x * 256 + threadIdx.x;
    if (idx >= MTOK * EDIM) return;
    int row = idx >> 8, e = idx & 255;
    int b = row / NTOK, l = row % NTOK;
    float v = (l < NPATCH) ? ptok[((size_t)(b * NPATCH + l)) * EDIM + e] : cls[e];
    tok[idx] = v + pos[l * EDIM + e];
}

__global__ void k_mean(const float* __restrict__ tok, float* __restrict__ g) {
    int idx = blockIdx.x * 256 + threadIdx.x;
    if (idx >= BATCH * EDIM) return;
    int b = idx >> 8, e = idx & 255;
    float s = 0.f;
    for (int l = 0; l < NTOK; ++l) s += tok[((size_t)(b * NTOK + l)) * EDIM + e];
    g[idx] = s * (1.f / 197.f);
}

__global__ __launch_bounds__(64) void k_router(const float* __restrict__ gmean,
    const float* __restrict__ r1w, const float* __restrict__ r1b,
    const float* __restrict__ r2w, const float* __restrict__ r2b,
    float* __restrict__ wsum) {
    __shared__ float hs[64];
    __shared__ float l4[4];
    int b = blockIdx.x, t = threadIdx.x;
    float a = r1b[t];
    for (int e = 0; e < EDIM; ++e) a += gmean[b * EDIM + e] * r1w[t * EDIM + e];
    hs[t] = gelu_f(a);
    __syncthreads();
    if (t < 4) {
        float a2 = r2b[t];
        for (int j = 0; j < 64; ++j) a2 += hs[j] * r2w[t * 64 + j];
        l4[t] = a2;
    }
    __syncthreads();
    if (t == 0) {
        float mx = fmaxf(fmaxf(l4[0], l4[1]), fmaxf(l4[2], l4[3]));
        float e0 = __expf(l4[0]-mx), e1 = __expf(l4[1]-mx), e2 = __expf(l4[2]-mx), e3 = __expf(l4[3]-mx);
        float inv = 1.f / (e0 + e1 + e2 + e3);
        // PERMS: N=197 not a square -> dirs {0,1} = identity, {2,3} = reversed
        wsum[0 * 8 + b] = (e0 + e1) * inv;
        wsum[1 * 8 + b] = (e2 + e3) * inv;
    }
}

// Expand per-batch router weight into a per-row scale vector (kills the
// integer division in the GEMM epilogue).
__global__ void k_expand_scale(const float* __restrict__ wsum, int g,
                               float* __restrict__ wscale) {
    int idx = blockIdx.x * 256 + threadIdx.x;
    if (idx >= MTOK) return;
    wscale[idx] = wsum[g * 8 + idx / NTOK];
}

__global__ __launch_bounds__(256) void k_layernorm(const float* __restrict__ in,
    float* __restrict__ out, const float* __restrict__ w, const float* __restrict__ bb) {
    __shared__ float red[256];
    int row = blockIdx.x, e = threadIdx.x;
    float v = in[(size_t)row * EDIM + e];
    red[e] = v; __syncthreads();
    for (int s = 128; s > 0; s >>= 1) { if (e < s) red[e] += red[e + s]; __syncthreads(); }
    float mu = red[0] * (1.f / 256.f);
    __syncthreads();
    float dv = v - mu;
    red[e] = dv * dv; __syncthreads();
    for (int s = 128; s > 0; s >>= 1) { if (e < s) red[e] += red[e + s]; __syncthreads(); }
    float var = red[0] * (1.f / 256.f);
    out[(size_t)row * EDIM + e] = dv * rsqrtf(var + 1e-5f) * w[e] + bb[e];
}

__global__ void k_reverse(const float* __restrict__ xn, float* __restrict__ xr) {
    int idx = blockIdx.x * 256 + threadIdx.x;
    if (idx >= MTOK * EDIM) return;
    int row = idx >> 8, e = idx & 255;
    int b = row / NTOK, l = row % NTOK;
    xr[idx] = xn[((size_t)(b * NTOK + (196 - l))) * EDIM + e];
}

__global__ void k_conv_silu(const float* __restrict__ xz, const float* __restrict__ cw,
                            const float* __restrict__ cb, float* __restrict__ xc) {
    int idx = blockIdx.x * 256 + threadIdx.x;
    if (idx >= MTOK * DI) return;
    int m = idx >> 9, d = idx & 511;
    int l = m % NTOK;
    float a = cb[d];
    #pragma unroll
    for (int kk = 0; kk < 4; ++kk) {
        int ll = l - 3 + kk;
        if (ll >= 0) a += xz[((size_t)(m - 3 + kk)) * 1024 + d] * cw[d * 4 + kk];
    }
    xc[idx] = silu_f(a);
}

// One wave32 per (batch, channel): 2 SSM states per lane, shuffle-reduced dot.
__global__ __launch_bounds__(256) void k_scan(
    const float* __restrict__ xc, const float* __restrict__ dtb,
    const float* __restrict__ proj, const float* __restrict__ xz,
    const float* __restrict__ A_log, const float* __restrict__ Dvec,
    float* __restrict__ y) {
    int wid  = (blockIdx.x * blockDim.x + threadIdx.x) >> 5;
    int lane = threadIdx.x & 31;
    int b = wid >> 9, d = wid & 511;
    int s1 = lane, s2 = lane + 32;
    float A1 = -__expf(A_log[d * 64 + s1]);
    float A2 = -__expf(A_log[d * 64 + s2]);
    float Dd = Dvec[d];
    float h1 = 0.f, h2 = 0.f;
    for (int l = 0; l < NTOK; ++l) {
        int m = b * NTOK + l;
        float dtv = dtb[(size_t)m * DI + d];
        float u   = xc[(size_t)m * DI + d];
        float Bv1 = proj[(size_t)m * 144 + DTRANK + s1];
        float Bv2 = proj[(size_t)m * 144 + DTRANK + s2];
        float Cv1 = proj[(size_t)m * 144 + DTRANK + DSTATE + s1];
        float Cv2 = proj[(size_t)m * 144 + DTRANK + DSTATE + s2];
        float du = dtv * u;
        h1 = h1 * __expf(dtv * A1) + du * Bv1;
        h2 = h2 * __expf(dtv * A2) + du * Bv2;
        float p = h1 * Cv1 + h2 * Cv2;
        #pragma unroll
        for (int off = 16; off > 0; off >>= 1) p += __shfl_xor(p, off, 32);
        if (lane == 0) {
            float zv = xz[(size_t)m * 1024 + DI + d];
            y[(size_t)m * DI + d] = (p + u * Dd) * silu_f(zv);
        }
    }
}

__global__ void k_xpad(const float* __restrict__ xn2, float* __restrict__ xpad) {
    int idx = blockIdx.x * 256 + threadIdx.x;
    if (idx >= MPAD * EDIM) return;
    int row = idx >> 8, e = idx & 255;
    int b = row / NPAD, l = row % NPAD;
    xpad[idx] = (l < NTOK) ? xn2[((size_t)(b * NTOK + l)) * EDIM + e] : 0.f;
}

// One wave32 per (window, head): WIN=4 tokens, dh=64 (2 dims per lane).
__global__ __launch_bounds__(128) void k_attn(const float* __restrict__ qkv,
                                              float* __restrict__ o) {
    int head = threadIdx.x >> 5;
    int lane = threadIdx.x & 31;
    int b = blockIdx.x / 50, win = blockIdx.x % 50;
    int row0 = b * NPAD + win * 4;
    float q1[4], q2[4], kk1[4], kk2[4], v1[4], v2[4];
    #pragma unroll
    for (int i = 0; i < 4; ++i) {
        size_t base = (size_t)(row0 + i) * 768 + head * 64;
        q1[i]  = qkv[base + lane];        q2[i]  = qkv[base + lane + 32];
        kk1[i] = qkv[base + 256 + lane];  kk2[i] = qkv[base + 256 + lane + 32];
        v1[i]  = qkv[base + 512 + lane];  v2[i]  = qkv[base + 512 + lane + 32];
    }
    float s[4][4];
    #pragma unroll
    for (int i = 0; i < 4; ++i)
        #pragma unroll
        for (int j = 0; j < 4; ++j) {
            float p = q1[i] * kk1[j] + q2[i] * kk2[j];
            #pragma unroll
            for (int off = 16; off > 0; off >>= 1) p += __shfl_xor(p, off, 32);
            s[i][j] = p * 0.125f;   // 1/sqrt(64)
        }
    #pragma unroll
    for (int i = 0; i < 4; ++i) {
        float mx = fmaxf(fmaxf(s[i][0], s[i][1]), fmaxf(s[i][2], s[i][3]));
        float e0 = __expf(s[i][0]-mx), e1 = __expf(s[i][1]-mx);
        float e2 = __expf(s[i][2]-mx), e3 = __expf(s[i][3]-mx);
        float inv = 1.f / (e0 + e1 + e2 + e3);
        float a0 = e0*inv, a1 = e1*inv, a2 = e2*inv, a3 = e3*inv;
        float o1 = a0*v1[0] + a1*v1[1] + a2*v1[2] + a3*v1[3];
        float o2 = a0*v2[0] + a1*v2[1] + a2*v2[2] + a3*v2[3];
        size_t ob = (size_t)(row0 + i) * EDIM + head * 64;
        o[ob + lane] = o1; o[ob + lane + 32] = o2;
    }
}

__global__ __launch_bounds__(256) void k_gate_residual_ln(
    const float* __restrict__ xn2, const float* __restrict__ o2,
    const float* __restrict__ gsc, const float* __restrict__ w,
    const float* __restrict__ bb, float* __restrict__ out) {
    __shared__ float red[256];
    int row = blockIdx.x, e = threadIdx.x;
    int b = row / NTOK, l = row % NTOK;
    float v = xn2[(size_t)row * EDIM + e] + gsc[0] * o2[((size_t)(b * NPAD + l)) * EDIM + e];
    red[e] = v; __syncthreads();
    for (int s = 128; s > 0; s >>= 1) { if (e < s) red[e] += red[e + s]; __syncthreads(); }
    float mu = red[0] * (1.f / 256.f);
    __syncthreads();
    float dv = v - mu;
    red[e] = dv * dv; __syncthreads();
    for (int s = 128; s > 0; s >>= 1) { if (e < s) red[e] += red[e + s]; __syncthreads(); }
    float var = red[0] * (1.f / 256.f);
    out[(size_t)row * EDIM + e] = dv * rsqrtf(var + 1e-5f) * w[e] + bb[e];
}

__global__ __launch_bounds__(128) void k_heads(const float* __restrict__ tokN,
    const float* __restrict__ c1w, const float* __restrict__ c1b,
    const float* __restrict__ c2w, const float* __restrict__ c2b,
    const float* __restrict__ r1w, const float* __restrict__ r1b,
    const float* __restrict__ r2w, const float* __restrict__ r2b,
    float* __restrict__ out) {
    __shared__ float cs[256], h1s[128], h2s[128];
    int b = blockIdx.x, t = threadIdx.x;
    const float* crow = tokN + ((size_t)(b * NTOK + 196)) * EDIM;
    cs[t] = crow[t]; cs[t + 128] = crow[t + 128];
    __syncthreads();
    float a1 = c1b[t], a2 = r1b[t];
    for (int e = 0; e < EDIM; ++e) {
        a1 += cs[e] * c1w[t * EDIM + e];
        a2 += cs[e] * r1w[t * EDIM + e];
    }
    h1s[t] = gelu_f(a1); h2s[t] = gelu_f(a2);
    __syncthreads();
    if (t < 2) { float a = c2b[t]; for (int j = 0; j < 128; ++j) a += h1s[j] * c2w[t * 128 + j]; out[b * 2 + t] = a; }
    if (t == 2){ float a = r2b[0]; for (int j = 0; j < 128; ++j) a += h2s[j] * r2w[j];          out[16 + b] = a; }
}

// ---------------------------------------------------------------------------
// Host orchestration (graph-capture safe: only launches on `stream`)
// ---------------------------------------------------------------------------
static inline void launch_gemm(const float* A, int lda, const float* B, const float* bias,
                               float* C, int ldc, int M, int N, int K, int act, int accum,
                               const float* rowScale, hipStream_t stream) {
    dim3 grid((N + GBN - 1) / GBN, (M + GBM - 1) / GBM);
    gemm_wmma<<<grid, 128, 0, stream>>>(A, lda, B, bias, C, ldc, M, N, K, act, accum, rowScale);
}

extern "C" void kernel_launch(void* const* d_in, const int* in_sizes, int n_in,
                              void* d_out, int out_size, void* d_ws, size_t ws_size,
                              hipStream_t stream) {
    auto F = [&](int i) { return (const float*)d_in[i]; };
    const float* x        = F(0);
    const float* colour_w = F(1);
    const float* patch_w  = F(2);
    const float* patch_b  = F(3);
    const float* cls_tok  = F(4);
    const float* pos_emb  = F(5);
    const float* norm_w   = F(102);
    const float* norm_b   = F(103);

    float* p = (float*)d_ws;
    float* weff   = p; p += (size_t)EDIM * 768;
    float* colb   = p; p += (size_t)MPATCH * 768;
    float* ptok   = p; p += (size_t)MPATCH * EDIM;
    float* tok    = p; p += (size_t)MTOK * EDIM;
    float* xn     = p; p += (size_t)MTOK * EDIM;
    float* xn2    = p; p += (size_t)MTOK * EDIM;
    float* xrev   = p; p += (size_t)MTOK * EDIM;
    float* gmean  = p; p += (size_t)BATCH * EDIM;
    float* wsum   = p; p += 16;
    float* wscale = p; p += (size_t)MTOK;
    float* xz     = p; p += (size_t)MTOK * 1024;
    float* xc     = p; p += (size_t)MTOK * DI;
    float* proj   = p; p += (size_t)MTOK * 144;
    float* dtb    = p; p += (size_t)MTOK * DI;
    float* ybuf   = p; p += (size_t)MTOK * DI;
    float* xpad   = p; p += (size_t)MPAD * EDIM;
    float* qkvb   = p; p += (size_t)MPAD * 768;
    float* obuf   = p; p += (size_t)MPAD * EDIM;
    float* o2     = p; p += (size_t)MPAD * EDIM;

    // ---- patch embedding: fold colour_w into patch_w, im2col, one WMMA GEMM
    k_weff<<<(EDIM * 768 + 255) / 256, 256, 0, stream>>>(patch_w, colour_w, weff);
    k_im2col<<<(MPATCH * 768 + 255) / 256, 256, 0, stream>>>(x, colb);
    launch_gemm(colb, 768, weff, patch_b, ptok, EDIM, MPATCH, EDIM, 768, 0, 0, nullptr, stream);
    k_tok_assemble<<<(MTOK * EDIM + 255) / 256, 256, 0, stream>>>(ptok, cls_tok, pos_emb, tok);

    for (int blk = 0; blk < 4; ++blk) {
        const int pb = 6 + 24 * blk;
        const float* r1w = F(pb+0);  const float* r1b = F(pb+1);
        const float* r2w = F(pb+2);  const float* r2b = F(pb+3);
        const float* n1w = F(pb+4);  const float* n1b = F(pb+5);
        const float* n2w = F(pb+6);  const float* n2b = F(pb+7);
        const float* in_w   = F(pb+8);
        const float* conv_w = F(pb+9);   const float* conv_b = F(pb+10);
        const float* x_w    = F(pb+11);
        const float* dt_w   = F(pb+12);  const float* dt_b   = F(pb+13);
        const float* A_log  = F(pb+14);  const float* Dvec   = F(pb+15);
        const float* out_w  = F(pb+16);
        const float* qkv_w  = F(pb+17);  const float* qkv_b  = F(pb+18);
        const float* gow    = F(pb+19);  const float* gob    = F(pb+20);
        const float* glnw   = F(pb+21);  const float* glnb   = F(pb+22);
        const float* gscale = F(pb+23);

        // router (PERMS dedupe: dirs {0,1} identity, {2,3} reverse)
        k_mean<<<(BATCH * EDIM + 255) / 256, 256, 0, stream>>>(tok, gmean);
        k_router<<<BATCH, 64, 0, stream>>>(gmean, r1w, r1b, r2w, r2b, wsum);
        k_layernorm<<<MTOK, 256, 0, stream>>>(tok, xn, n1w, n1b);
        k_reverse<<<(MTOK * EDIM + 255) / 256, 256, 0, stream>>>(xn, xrev);

        for (int g = 0; g < 2; ++g) {
            const float* Ain = (g == 0) ? xn : xrev;
            // in_proj: [1576,256] x [1024,256]^T
            launch_gemm(Ain, EDIM, in_w, nullptr, xz, 1024, MTOK, 1024, EDIM, 0, 0, nullptr, stream);
            k_conv_silu<<<(MTOK * DI + 255) / 256, 256, 0, stream>>>(xz, conv_w, conv_b, xc);
            // x_proj: [1576,512] x [144,512]^T
            launch_gemm(xc, DI, x_w, nullptr, proj, 144, MTOK, 144, DI, 0, 0, nullptr, stream);
            // dt: softplus(proj[:, :16] x [512,16]^T + dt_b)
            launch_gemm(proj, 144, dt_w, dt_b, dtb, DI, MTOK, DI, DTRANK, 3, 0, nullptr, stream);
            // selective scan: 4096 wave32's (one per (b,d))
            k_scan<<<512, 256, 0, stream>>>(xc, dtb, proj, xz, A_log, Dvec, ybuf);
            // out_proj, router-weighted, accumulated into tok residual
            k_expand_scale<<<(MTOK + 255) / 256, 256, 0, stream>>>(wsum, g, wscale);
            launch_gemm(ybuf, DI, out_w, nullptr, tok, EDIM, MTOK, EDIM, DI, 0, 1, wscale, stream);
        }

        // gated windowed attention
        k_layernorm<<<MTOK, 256, 0, stream>>>(tok, xn2, n2w, n2b);
        k_xpad<<<(MPAD * EDIM + 255) / 256, 256, 0, stream>>>(xn2, xpad);
        launch_gemm(xpad, EDIM, qkv_w, qkv_b, qkvb, 768, MPAD, 768, EDIM, 0, 0, nullptr, stream);
        k_attn<<<BATCH * 50, 128, 0, stream>>>(qkvb, obuf);
        launch_gemm(obuf, EDIM, gow, gob, o2, EDIM, MPAD, EDIM, EDIM, 0, 0, nullptr, stream);
        k_gate_residual_ln<<<MTOK, 256, 0, stream>>>(xn2, o2, gscale, glnw, glnb, tok);
    }

    // final norm + heads
    k_layernorm<<<MTOK, 256, 0, stream>>>(tok, xn, norm_w, norm_b);
    k_heads<<<BATCH, 128, 0, stream>>>(xn, F(104), F(105), F(106), F(107),
                                       F(108), F(109), F(110), F(111), (float*)d_out);
}
